// Encoder_21311627723475
// MI455X (gfx1250) — compile-verified, matrix operand order checked by reading, not run on hardware
//
#include <hip/hip_runtime.h>
#include <hip/hip_bf16.h>
#include <math.h>
#include <stdint.h>

// ---------------------------------------------------------------------------
// Sizes (from reference)
// ---------------------------------------------------------------------------
#define S_TOT 2560      // B*N sentences
#define T_LEN 70
#define HID   200
#define RELN  100
#define BATCH 128
#define NBAG  20
#define IN_D  60
#define NREAL 600       // 3*HID
#define NPAD  640       // padded to multiple of 64 for 16x64 N-groups
#define KIH   64        // IN_D padded to multiple of 32
#define KHH   224       // HID padded to multiple of 32
#define KLPAD 8         // LDS row padding (bf16 elems) to avoid bank conflicts

typedef __bf16 bf16_t;
typedef bf16_t bf16x16 __attribute__((ext_vector_type(16)));
typedef bf16_t bf16x8  __attribute__((ext_vector_type(8)));
typedef float  f32x8   __attribute__((ext_vector_type(8)));

// ---------------------------------------------------------------------------
// Pack word+pos embeddings into bf16 x[S,T,KIH] (zero-padded 60->64)
// ---------------------------------------------------------------------------
__global__ __launch_bounds__(256)
void embed_pack_kernel(const int* __restrict__ tok, const int* __restrict__ p1,
                       const int* __restrict__ p2, const float* __restrict__ emb,
                       const float* __restrict__ pemb, bf16_t* __restrict__ xb)
{
    long long idx = (long long)blockIdx.x * 256 + threadIdx.x;
    const long long total = (long long)S_TOT * T_LEN * KIH;
    if (idx >= total) return;
    int pos = (int)(idx >> 6);
    int k   = (int)(idx & 63);
    float v = 0.f;
    if (k < 50)       v = emb[(size_t)tok[pos] * 50 + k];
    else if (k < 55)  v = pemb[p1[pos] * 5 + (k - 50)];
    else if (k < 60)  v = pemb[p2[pos] * 5 + (k - 55)];
    xb[idx] = (bf16_t)v;
}

// ---------------------------------------------------------------------------
// Pack fp32 weight [600,Kd] -> bf16 [NPAD,Kp], zero padded rows/cols
// ---------------------------------------------------------------------------
__global__ __launch_bounds__(256)
void pack_w_kernel(const float* __restrict__ W, bf16_t* __restrict__ Wb,
                   int Kd, int Kp)
{
    int idx = blockIdx.x * 256 + threadIdx.x;
    int total = NPAD * Kp;
    if (idx >= total) return;
    int n = idx / Kp, k = idx % Kp;
    float v = (n < NREAL && k < Kd) ? W[n * Kd + k] : 0.f;
    Wb[idx] = (bf16_t)v;
}

// ---------------------------------------------------------------------------
// Zero hidden state (fp32 [S,HID] and bf16 padded [S,KHH])
// ---------------------------------------------------------------------------
__global__ __launch_bounds__(256)
void init_h_kernel(float* __restrict__ h, bf16_t* __restrict__ hb)
{
    int idx = blockIdx.x * 256 + threadIdx.x;
    if (idx >= S_TOT * KHH) return;
    int s = idx / KHH, d = idx % KHH;
    hb[idx] = (bf16_t)0.f;
    if (d < HID) h[s * HID + d] = 0.f;
}

// ---------------------------------------------------------------------------
// WMMA GEMM: C[M, ldc] = A[M,Kp](bf16) * B[NPAD,Kp](bf16)^T + bias
//
// Block (8 waves) computes a 256x64 tile of C. The 64xKp weight slice shared
// by all 8 waves is DMA'd into LDS once with global_load_async_to_lds_b128
// (ASYNCcnt-tracked, no VGPR round trip), then each wave computes a 32x64
// subtile: 2 A-fragments (global) x 4 B-fragments (LDS, ds_load_b128) ->
// 8x v_wmma_f32_16x16x32_bf16 per 32-wide K-step.
//
// Fragment layout per ISA 7.12.2 (16-bit A 16x32): lane holds row (lane&15),
// K runs [kbase..kbase+7] and [kbase+16..kbase+23], kbase=(lane>>4)*8.
// LDS rows padded by KLPAD bf16 so fragment reads spread across banks.
// ---------------------------------------------------------------------------
__global__ __launch_bounds__(256)
void wmma_gemm_bf16_nt(const bf16_t* __restrict__ A, int lda,
                       const bf16_t* __restrict__ B, int ldb,   // ldb == Kp
                       const float* __restrict__ bias, int n_real,
                       float* __restrict__ C, int ldc,
                       int Kp, int ntile4)
{
    __shared__ bf16_t Btile[64 * (KHH + KLPAD)];
    const int kld = Kp + KLPAD;

    int tid  = threadIdx.x;
    int blkM = blockIdx.x / ntile4;
    int tN4  = blockIdx.x % ntile4;

    // ---- async DMA stage: B[tN4*64 .. +63][0..Kp) -> LDS (padded rows) ----
    {
        const int cpr     = Kp >> 3;      // 16B chunks per row
        const int nchunks = 64 * cpr;     // 512 (KIH) / 1792 (KHH): % 256 == 0
        const bf16_t* Bsrc = B + (size_t)(tN4 * 64) * ldb;
        unsigned lbase = (unsigned)(uintptr_t)&Btile[0];   // LDS offset = addr[31:0]
        for (int c = tid; c < nchunks; c += 256) {
            int row = c / cpr, col = c - row * cpr;
            unsigned laddr = lbase + (unsigned)(row * kld + col * 8) * 2u;
            const bf16_t* gaddr = Bsrc + (size_t)row * ldb + col * 8;
            asm volatile("global_load_async_to_lds_b128 %0, %1, off"
                         :: "v"(laddr), "v"(gaddr) : "memory");
        }
        asm volatile("s_wait_asynccnt 0x0" ::: "memory");
    }
    __syncthreads();

    int lane = tid & 31;
    int wv   = tid >> 5;                  // 0..7 -> M-subrange within block
    int half = lane >> 4;                 // 0 or 1
    int l16  = lane & 15;
    int koff = half * 8;

    const bf16_t* Arow0 = A + (size_t)(blkM * 256 + wv * 32 + l16) * lda;
    const bf16_t* Arow1 = Arow0 + (size_t)16 * lda;
    const bf16_t* L0 = &Btile[(size_t)( 0 + l16) * kld];
    const bf16_t* L1 = &Btile[(size_t)(16 + l16) * kld];
    const bf16_t* L2 = &Btile[(size_t)(32 + l16) * kld];
    const bf16_t* L3 = &Btile[(size_t)(48 + l16) * kld];

    f32x8 acc00 = {}; f32x8 acc01 = {}; f32x8 acc02 = {}; f32x8 acc03 = {};
    f32x8 acc10 = {}; f32x8 acc11 = {}; f32x8 acc12 = {}; f32x8 acc13 = {};

    for (int k0 = 0; k0 < Kp; k0 += 32) {
        bf16x8 a0l = *(const bf16x8*)(Arow0 + k0 + koff);
        bf16x8 a0h = *(const bf16x8*)(Arow0 + k0 + 16 + koff);
        bf16x16 a0 = __builtin_shufflevector(a0l, a0h,
                        0,1,2,3,4,5,6,7,8,9,10,11,12,13,14,15);
        bf16x8 a1l = *(const bf16x8*)(Arow1 + k0 + koff);
        bf16x8 a1h = *(const bf16x8*)(Arow1 + k0 + 16 + koff);
        bf16x16 a1 = __builtin_shufflevector(a1l, a1h,
                        0,1,2,3,4,5,6,7,8,9,10,11,12,13,14,15);

        bf16x8 b0l = *(const bf16x8*)(L0 + k0 + koff);
        bf16x8 b0h = *(const bf16x8*)(L0 + k0 + 16 + koff);
        bf16x16 b0 = __builtin_shufflevector(b0l, b0h,
                        0,1,2,3,4,5,6,7,8,9,10,11,12,13,14,15);
        bf16x8 b1l = *(const bf16x8*)(L1 + k0 + koff);
        bf16x8 b1h = *(const bf16x8*)(L1 + k0 + 16 + koff);
        bf16x16 b1 = __builtin_shufflevector(b1l, b1h,
                        0,1,2,3,4,5,6,7,8,9,10,11,12,13,14,15);
        bf16x8 b2l = *(const bf16x8*)(L2 + k0 + koff);
        bf16x8 b2h = *(const bf16x8*)(L2 + k0 + 16 + koff);
        bf16x16 b2 = __builtin_shufflevector(b2l, b2h,
                        0,1,2,3,4,5,6,7,8,9,10,11,12,13,14,15);
        bf16x8 b3l = *(const bf16x8*)(L3 + k0 + koff);
        bf16x8 b3h = *(const bf16x8*)(L3 + k0 + 16 + koff);
        bf16x16 b3 = __builtin_shufflevector(b3l, b3h,
                        0,1,2,3,4,5,6,7,8,9,10,11,12,13,14,15);

        acc00 = __builtin_amdgcn_wmma_f32_16x16x32_bf16(false, a0, false, b0,
                                                        (short)0, acc00, false, false);
        acc01 = __builtin_amdgcn_wmma_f32_16x16x32_bf16(false, a0, false, b1,
                                                        (short)0, acc01, false, false);
        acc02 = __builtin_amdgcn_wmma_f32_16x16x32_bf16(false, a0, false, b2,
                                                        (short)0, acc02, false, false);
        acc03 = __builtin_amdgcn_wmma_f32_16x16x32_bf16(false, a0, false, b3,
                                                        (short)0, acc03, false, false);
        acc10 = __builtin_amdgcn_wmma_f32_16x16x32_bf16(false, a1, false, b0,
                                                        (short)0, acc10, false, false);
        acc11 = __builtin_amdgcn_wmma_f32_16x16x32_bf16(false, a1, false, b1,
                                                        (short)0, acc11, false, false);
        acc12 = __builtin_amdgcn_wmma_f32_16x16x32_bf16(false, a1, false, b2,
                                                        (short)0, acc12, false, false);
        acc13 = __builtin_amdgcn_wmma_f32_16x16x32_bf16(false, a1, false, b3,
                                                        (short)0, acc13, false, false);
    }

    // C/D layout: element i -> row (subtile base + i + 8*half), col l16
    int c0 = tN4 * 64 +  0 + l16;
    int c1 = tN4 * 64 + 16 + l16;
    int c2 = tN4 * 64 + 32 + l16;
    int c3 = tN4 * 64 + 48 + l16;
    float bv0 = (bias && c0 < n_real) ? bias[c0] : 0.f;
    float bv1 = (bias && c1 < n_real) ? bias[c1] : 0.f;
    float bv2 = (bias && c2 < n_real) ? bias[c2] : 0.f;
    float bv3 = (bias && c3 < n_real) ? bias[c3] : 0.f;
    int rbase = blkM * 256 + wv * 32 + half * 8;
    #pragma unroll
    for (int i = 0; i < 8; ++i) {
        float* Crow0 = C + (size_t)(rbase + i) * ldc;
        float* Crow1 = C + (size_t)(rbase + 16 + i) * ldc;
        Crow0[c0] = acc00[i] + bv0;
        Crow0[c1] = acc01[i] + bv1;
        Crow0[c2] = acc02[i] + bv2;
        Crow0[c3] = acc03[i] + bv3;
        Crow1[c0] = acc10[i] + bv0;
        Crow1[c1] = acc11[i] + bv1;
        Crow1[c2] = acc12[i] + bv2;
        Crow1[c3] = acc13[i] + bv3;
    }
}

// ---------------------------------------------------------------------------
// GRU gate math: g1 = x_t@W_ih^T+b_ih, g2 = h@W_hh^T+b_hh (both [S,NPAD]).
// Updates h (fp32), hb (bf16 padded), and tup[s, t_out, :] (= or +=).
// ---------------------------------------------------------------------------
__global__ __launch_bounds__(256)
void gru_gate_kernel(const float* __restrict__ g1, const float* __restrict__ g2,
                     float* __restrict__ h, bf16_t* __restrict__ hb,
                     float* __restrict__ tup, int t_out, int accumulate)
{
    int idx = blockIdx.x * 256 + threadIdx.x;
    if (idx >= S_TOT * HID) return;
    int s = idx / HID, d = idx % HID;
    const float* G1 = g1 + (size_t)s * NPAD;
    const float* G2 = g2 + (size_t)s * NPAD;
    float xr = G1[d], xz = G1[HID + d], xn = G1[2 * HID + d];
    float hr = G2[d], hz = G2[HID + d], hn = G2[2 * HID + d];
    float r = 1.f / (1.f + __expf(-(xr + hr)));
    float z = 1.f / (1.f + __expf(-(xz + hz)));
    float n = tanhf(xn + r * hn);
    float hp = h[idx];
    float hv = (1.f - z) * n + z * hp;
    h[idx] = hv;
    hb[(size_t)s * KHH + d] = (bf16_t)hv;
    float* tp = tup + ((size_t)s * T_LEN + t_out) * HID + d;
    if (accumulate) *tp += hv; else *tp = hv;
}

// ---------------------------------------------------------------------------
// Word-level attention over T per sentence
// ---------------------------------------------------------------------------
__global__ __launch_bounds__(256)
void word_attn_kernel(const float* __restrict__ tup, const float* __restrict__ aw,
                      float* __restrict__ H)
{
    int s = blockIdx.x;
    __shared__ float sc[T_LEN];
    __shared__ float alpha[T_LEN];
    const float* base = tup + (size_t)s * T_LEN * HID;
    for (int t = threadIdx.x; t < T_LEN; t += blockDim.x) {
        float acc = 0.f;
        for (int d = 0; d < HID; ++d) acc += tanhf(base[t * HID + d]) * aw[d];
        sc[t] = acc;
    }
    __syncthreads();
    if (threadIdx.x == 0) {
        float m = sc[0];
        for (int t = 1; t < T_LEN; ++t) m = fmaxf(m, sc[t]);
        float sum = 0.f;
        for (int t = 0; t < T_LEN; ++t) { float e = __expf(sc[t] - m); alpha[t] = e; sum += e; }
        float inv = 1.f / sum;
        for (int t = 0; t < T_LEN; ++t) alpha[t] *= inv;
    }
    __syncthreads();
    for (int d = threadIdx.x; d < HID; d += blockDim.x) {
        float acc = 0.f;
        for (int t = 0; t < T_LEN; ++t) acc += alpha[t] * base[t * HID + d];
        H[(size_t)s * HID + d] = acc;
    }
}

// ---------------------------------------------------------------------------
// Bag-level attention over N per bag
// ---------------------------------------------------------------------------
__global__ __launch_bounds__(256)
void bag_attn_kernel(const float* __restrict__ H, const float* __restrict__ sen_a,
                     const float* __restrict__ sen_r, float* __restrict__ Sv)
{
    int b = blockIdx.x;
    __shared__ float Aw[NBAG];
    const float* Hb = H + (size_t)b * NBAG * HID;
    if (threadIdx.x < NBAG) {
        float acc = 0.f;
        for (int d = 0; d < HID; ++d)
            acc += Hb[threadIdx.x * HID + d] * sen_a[d] * sen_r[d];
        Aw[threadIdx.x] = acc;
    }
    __syncthreads();
    if (threadIdx.x == 0) {
        float m = Aw[0];
        for (int n = 1; n < NBAG; ++n) m = fmaxf(m, Aw[n]);
        float sum = 0.f;
        for (int n = 0; n < NBAG; ++n) { float e = __expf(Aw[n] - m); Aw[n] = e; sum += e; }
        float inv = 1.f / sum;
        for (int n = 0; n < NBAG; ++n) Aw[n] *= inv;
    }
    __syncthreads();
    for (int d = threadIdx.x; d < HID; d += blockDim.x) {
        float acc = 0.f;
        for (int n = 0; n < NBAG; ++n) acc += Aw[n] * Hb[n * HID + d];
        Sv[(size_t)b * HID + d] = acc;
    }
}

// ---------------------------------------------------------------------------
// logits[b,r] = Sv[b,:] . relemb[r,:] + sen_d[r]
// ---------------------------------------------------------------------------
__global__ __launch_bounds__(256)
void logits_kernel(const float* __restrict__ Sv, const float* __restrict__ relemb,
                   const float* __restrict__ sen_d, float* __restrict__ logits)
{
    int idx = blockIdx.x * 256 + threadIdx.x;
    if (idx >= BATCH * RELN) return;
    int b = idx / RELN, r = idx % RELN;
    float acc = sen_d[r];
    const float* sv = Sv + (size_t)b * HID;
    const float* re = relemb + (size_t)r * HID;
    for (int d = 0; d < HID; ++d) acc += sv[d] * re[d];
    logits[idx] = acc;
}

// ---------------------------------------------------------------------------
// total_loss = sum_b mean_r BCEWithLogits
// ---------------------------------------------------------------------------
__global__ __launch_bounds__(256)
void loss_kernel(const float* __restrict__ logits, const int* __restrict__ labels,
                 float* __restrict__ out_loss)
{
    __shared__ float red[256];
    float acc = 0.f;
    for (int idx = threadIdx.x; idx < BATCH * RELN; idx += 256) {
        int b = idx / RELN, r = idx % RELN;
        float l = logits[idx];
        float tgt = (labels[b] == r) ? 1.f : 0.f;
        acc += fmaxf(l, 0.f) - l * tgt + log1pf(__expf(-fabsf(l)));
    }
    red[threadIdx.x] = acc;
    __syncthreads();
    for (int s = 128; s > 0; s >>= 1) {
        if (threadIdx.x < s) red[threadIdx.x] += red[threadIdx.x + s];
        __syncthreads();
    }
    if (threadIdx.x == 0) out_loss[0] = red[0] / (float)RELN;
}

// ---------------------------------------------------------------------------
// Host orchestration
// ---------------------------------------------------------------------------
extern "C" void kernel_launch(void* const* d_in, const int* in_sizes, int n_in,
                              void* d_out, int out_size, void* d_ws, size_t ws_size,
                              hipStream_t stream)
{
    const int*   tok     = (const int*)d_in[0];
    const int*   p1      = (const int*)d_in[1];
    const int*   p2      = (const int*)d_in[2];
    const int*   labels  = (const int*)d_in[3];
    const float* emb     = (const float*)d_in[4];
    const float* pemb    = (const float*)d_in[5];
    const float* W_ih_f  = (const float*)d_in[6];
    const float* W_hh_f  = (const float*)d_in[7];
    const float* b_ih_f  = (const float*)d_in[8];
    const float* b_hh_f  = (const float*)d_in[9];
    const float* W_ih_b  = (const float*)d_in[10];
    const float* W_hh_b  = (const float*)d_in[11];
    const float* b_ih_b  = (const float*)d_in[12];
    const float* b_hh_b  = (const float*)d_in[13];
    const float* att_w   = (const float*)d_in[14];
    const float* sen_a   = (const float*)d_in[15];
    const float* sen_r   = (const float*)d_in[16];
    const float* relemb  = (const float*)d_in[17];
    const float* sen_d   = (const float*)d_in[18];

    // workspace carve-out (256B aligned)
    char* ws = (char*)d_ws;
    size_t off = 0;
    auto carve = [&](size_t bytes) -> void* {
        void* p = ws + off;
        off = (off + bytes + 255) & ~(size_t)255;
        return p;
    };
    bf16_t* xb    = (bf16_t*)carve((size_t)S_TOT * T_LEN * KIH * sizeof(bf16_t));
    bf16_t* Wihf  = (bf16_t*)carve((size_t)NPAD * KIH * sizeof(bf16_t));
    bf16_t* Whhf  = (bf16_t*)carve((size_t)NPAD * KHH * sizeof(bf16_t));
    bf16_t* Wihb  = (bf16_t*)carve((size_t)NPAD * KIH * sizeof(bf16_t));
    bf16_t* Whhb  = (bf16_t*)carve((size_t)NPAD * KHH * sizeof(bf16_t));
    float*  h     = (float*) carve((size_t)S_TOT * HID * sizeof(float));
    bf16_t* hb    = (bf16_t*)carve((size_t)S_TOT * KHH * sizeof(bf16_t));
    float*  g1    = (float*) carve((size_t)S_TOT * NPAD * sizeof(float));
    float*  g2    = (float*) carve((size_t)S_TOT * NPAD * sizeof(float));
    float*  tup   = (float*) carve((size_t)S_TOT * T_LEN * HID * sizeof(float));
    float*  Hbuf  = (float*) carve((size_t)S_TOT * HID * sizeof(float));
    float*  Sv    = (float*) carve((size_t)BATCH * HID * sizeof(float));

    float* out_loss   = (float*)d_out;
    float* out_logits = (float*)d_out + 1;

    // 1) pack inputs/weights to bf16
    {
        long long total = (long long)S_TOT * T_LEN * KIH;
        int blocks = (int)((total + 255) / 256);
        embed_pack_kernel<<<blocks, 256, 0, stream>>>(tok, p1, p2, emb, pemb, xb);
    }
    pack_w_kernel<<<(NPAD * KIH + 255) / 256, 256, 0, stream>>>(W_ih_f, Wihf, IN_D, KIH);
    pack_w_kernel<<<(NPAD * KHH + 255) / 256, 256, 0, stream>>>(W_hh_f, Whhf, HID, KHH);
    pack_w_kernel<<<(NPAD * KIH + 255) / 256, 256, 0, stream>>>(W_ih_b, Wihb, IN_D, KIH);
    pack_w_kernel<<<(NPAD * KHH + 255) / 256, 256, 0, stream>>>(W_hh_b, Whhb, HID, KHH);

    const int ntile4  = NPAD / 64;                       // 10
    const int gblocks = (S_TOT / 256) * ntile4;          // 100 blocks of 256
    const int gateblk = (S_TOT * HID + 255) / 256;
    const int initblk = (S_TOT * KHH + 255) / 256;
    const int ldx = T_LEN * KIH;                         // 4480

    // 2) forward GRU
    init_h_kernel<<<initblk, 256, 0, stream>>>(h, hb);
    for (int t = 0; t < T_LEN; ++t) {
        wmma_gemm_bf16_nt<<<gblocks, 256, 0, stream>>>(
            xb + (size_t)t * KIH, ldx, Wihf, KIH, b_ih_f, NREAL,
            g1, NPAD, KIH, ntile4);
        wmma_gemm_bf16_nt<<<gblocks, 256, 0, stream>>>(
            hb, KHH, Whhf, KHH, b_hh_f, NREAL,
            g2, NPAD, KHH, ntile4);
        gru_gate_kernel<<<gateblk, 256, 0, stream>>>(g1, g2, h, hb, tup, t, 0);
    }

    // 3) backward GRU (reads x time-reversed, accumulates into tup)
    init_h_kernel<<<initblk, 256, 0, stream>>>(h, hb);
    for (int t = 0; t < T_LEN; ++t) {
        int tin = T_LEN - 1 - t;
        wmma_gemm_bf16_nt<<<gblocks, 256, 0, stream>>>(
            xb + (size_t)tin * KIH, ldx, Wihb, KIH, b_ih_b, NREAL,
            g1, NPAD, KIH, ntile4);
        wmma_gemm_bf16_nt<<<gblocks, 256, 0, stream>>>(
            hb, KHH, Whhb, KHH, b_hh_b, NREAL,
            g2, NPAD, KHH, ntile4);
        gru_gate_kernel<<<gateblk, 256, 0, stream>>>(g1, g2, h, hb, tup, tin, 1);
    }

    // 4) attention + logits + loss
    word_attn_kernel<<<S_TOT, 256, 0, stream>>>(tup, att_w, Hbuf);
    bag_attn_kernel<<<BATCH, 256, 0, stream>>>(Hbuf, sen_a, sen_r, Sv);
    logits_kernel<<<(BATCH * RELN + 255) / 256, 256, 0, stream>>>(Sv, relemb, sen_d, out_logits);
    loss_kernel<<<1, 256, 0, stream>>>(out_logits, labels, out_loss);
}
